// myLstm_21234318311760
// MI455X (gfx1250) — compile-verified, matrix operand order checked by reading, not run on hardware
//
#include <hip/hip_runtime.h>
#include <hip/hip_bf16.h>

typedef __attribute__((ext_vector_type(16))) __bf16 v16bf;
typedef __attribute__((ext_vector_type(8)))  __bf16 v8bf;
typedef __attribute__((ext_vector_type(4)))  __bf16 v4bf;
typedef __attribute__((ext_vector_type(2)))  __bf16 v2bf;
typedef __attribute__((ext_vector_type(8)))  float  v8f;
typedef __attribute__((ext_vector_type(4)))  float  v4f;

#define B_ROWS 16384
#define DDIM   512
#define UDIM   512
#define KTOT   1024   // D + U fused reduction
#define BM     128    // rows per workgroup
#define BN     64     // u-columns per workgroup (z columns = 4*BN = 256)
#define BK     32     // K-step (one bf16 WMMA K)
#define NZ     256    // z columns per workgroup tile (4 gates x BN)
#define ASTRIDE 40    // bf16 elems per LDS row (padded from 32; 80B, 16B-multiple)
#define BSTRIDE 40

// Fallback kernel tiling (round-1 style, BM=64)
#define FBM    64

__device__ __forceinline__ float fast_sigmoid(float x) {
    return 1.0f / (1.0f + __expf(-x));
}
__device__ __forceinline__ float fast_tanh(float x) {
    return 1.0f - 2.0f / (__expf(2.0f * x) + 1.0f);
}

// Async 16B global->LDS copy. IOFFSET is added to BOTH the global and LDS
// address (ISA 10.7/15.18), so chunk offsets ride the immediate.
template <int OFF>
__device__ __forceinline__ void async_copy_b128_off(unsigned int lds_off, const void* gptr) {
    asm volatile("global_load_async_to_lds_b128 %0, %1, off offset:%2"
                 :: "v"(lds_off), "v"(gptr), "n"(OFF)
                 : "memory");
}
__device__ __forceinline__ void wait_async0() {
    asm volatile("s_wait_asynccnt 0" ::: "memory");
}

// ---------------------------------------------------------------------------
// Pack pass 1: X' = [inputs | h] as bf16, row-major [B_ROWS][KTOT]
// ---------------------------------------------------------------------------
__global__ __launch_bounds__(256)
void pack_x_kernel(const float* __restrict__ X, const float* __restrict__ H,
                   __bf16* __restrict__ Xbf) {
    const size_t i   = (size_t)blockIdx.x * 256 + threadIdx.x;   // quad tasks
    const size_t row = i >> 8;            // KTOT/4 = 256 quads per row
    const int    col = (int)(i & 255) * 4;
    const float* src = (col < DDIM) ? (X + row * DDIM + col)
                                    : (H + row * UDIM + (col - DDIM));
    v4f f = *(const v4f*)src;
    v4bf b;
    b.x = (__bf16)f.x; b.y = (__bf16)f.y; b.z = (__bf16)f.z; b.w = (__bf16)f.w;
    *(v4bf*)(Xbf + row * KTOT + col) = b;
}

// ---------------------------------------------------------------------------
// Pack pass 2: W'^T as bf16, [2048 zcols][KTOT k], zc = gate*512 + u.
// ---------------------------------------------------------------------------
__global__ __launch_bounds__(256)
void pack_w_kernel(const float* __restrict__ w_i, const float* __restrict__ u_i,
                   const float* __restrict__ w_f, const float* __restrict__ u_f,
                   const float* __restrict__ w_c, const float* __restrict__ u_c,
                   const float* __restrict__ w_o, const float* __restrict__ u_o,
                   __bf16* __restrict__ Wbf) {
    const size_t i  = (size_t)blockIdx.x * 256 + threadIdx.x;
    const int    zc = (int)(i >> 8);          // one zc row per block (uniform g)
    const int    k0 = (int)(i & 255) * 4;
    const int    g  = zc >> 9;
    const int    u  = zc & 511;
    const float* wp;
    if (k0 < DDIM) {
        const float* wg = (g == 0) ? w_i : (g == 1) ? w_f : (g == 2) ? w_c : w_o;
        wp = wg + (size_t)k0 * UDIM + u;
    } else {
        const float* ug = (g == 0) ? u_i : (g == 1) ? u_f : (g == 2) ? u_c : u_o;
        wp = ug + (size_t)(k0 - DDIM) * UDIM + u;
    }
    v4bf b;
    b.x = (__bf16)wp[0];
    b.y = (__bf16)wp[UDIM];
    b.z = (__bf16)wp[2 * UDIM];
    b.w = (__bf16)wp[3 * UDIM];
    *(v4bf*)(Wbf + (size_t)zc * KTOT + k0) = b;
}

// ---------------------------------------------------------------------------
// Main GEMM + fused LSTM epilogue. bf16 operands, async global->LDS staging.
// 128x(64u x 4gates) tile, 16 WMMAs per K-step per wave.
// ---------------------------------------------------------------------------
__global__ __launch_bounds__(256)
void lstm_wmma_async_kernel(const __bf16* __restrict__ Xbf,  // [B_ROWS][KTOT]
                            const __bf16* __restrict__ Wbf,  // [2048][KTOT]
                            const float* __restrict__ Cin,
                            const float* __restrict__ b_i, const float* __restrict__ b_f,
                            const float* __restrict__ b_c, const float* __restrict__ b_o,
                            float* __restrict__ Hout, float* __restrict__ Cout)
{
    const int tid   = threadIdx.x;
    const int wid   = tid >> 5;
    const int lane  = tid & 31;
    const int waveM = wid >> 2;          // 0..1 -> 64-row slab
    const int waveN = wid & 3;           // 0..3 -> 16 u-columns
    const int half  = lane >> 4;
    const int lidx  = lane & 15;

    const int blockRow = blockIdx.y * BM;
    const int uBase    = blockIdx.x * BN;

    __shared__ __align__(16) __bf16 sA[2][BM][ASTRIDE];
    __shared__ __align__(16) __bf16 sB[2][NZ][BSTRIDE];

    // ---- per-thread async-copy bases ----
    // A: row = tid&127, chunk pair = (tid>>7)*2 -> offsets 0/16 off one base
    const int aRow    = tid & 127;
    const int aChunkB = (tid >> 7) * 32;     // byte offset of chunk pair start
    const char* gA = (const char*)(Xbf + (size_t)(blockRow + aRow) * KTOT) + aChunkB;
    unsigned int ldsA[2];
    ldsA[0] = (unsigned int)(uintptr_t)&sA[0][aRow][0] + aChunkB;
    ldsA[1] = (unsigned int)(uintptr_t)&sA[1][aRow][0] + aChunkB;

    // B: zc-local column c = tid (0..255); 4 chunks -> offsets 0/16/32/48
    const int c    = tid;
    const int zrow = (c >> 6) * 512 + uBase + (c & 63);
    const char* gB = (const char*)(Wbf + (size_t)zrow * KTOT);
    unsigned int ldsB[2];
    ldsB[0] = (unsigned int)(uintptr_t)&sB[0][c][0];
    ldsB[1] = (unsigned int)(uintptr_t)&sB[1][c][0];

    auto issue_stage = [&](int buf) {
        async_copy_b128_off<0 >(ldsA[buf], gA);
        async_copy_b128_off<16>(ldsA[buf], gA);
        async_copy_b128_off<0 >(ldsB[buf], gB);
        async_copy_b128_off<16>(ldsB[buf], gB);
        async_copy_b128_off<32>(ldsB[buf], gB);
        async_copy_b128_off<48>(ldsB[buf], gB);
        gA += BK * 2;   // advance one K-step (64B)
        gB += BK * 2;
    };

    // ---- WMMA fragment loads (16-bit layout: lanes0-15 K0-7|16-23, lanes16-31 K8-15|24-31)
    auto loadA = [&](int buf, int mt) -> v16bf {
        const int row = waveM * 64 + mt * 16 + lidx;
        const __bf16* p = &sA[buf][row][half * 8];
        v8bf lo = *(const v8bf*)p;
        v8bf hi = *(const v8bf*)(p + 16);
        return __builtin_shufflevector(lo, hi, 0,1,2,3,4,5,6,7,8,9,10,11,12,13,14,15);
    };
    auto loadB = [&](int buf, int g) -> v16bf {
        const int row = g * 64 + waveN * 16 + lidx;
        const __bf16* p = &sB[buf][row][half * 8];
        v8bf lo = *(const v8bf*)p;
        v8bf hi = *(const v8bf*)(p + 16);
        return __builtin_shufflevector(lo, hi, 0,1,2,3,4,5,6,7,8,9,10,11,12,13,14,15);
    };

    v8f acc[4][4] = {};   // [m-tile][gate]

    issue_stage(0);
    wait_async0();
    __syncthreads();

    const int NK = KTOT / BK;   // 32
    for (int ks = 0; ks < NK; ++ks) {
        const int buf = ks & 1;
        if (ks + 1 < NK)
            issue_stage(buf ^ 1);   // async fill of the other buffer

        // Load all fragments first, then a dense 16-WMMA burst.
        v16bf a0 = loadA(buf, 0);
        v16bf a1 = loadA(buf, 1);
        v16bf a2 = loadA(buf, 2);
        v16bf a3 = loadA(buf, 3);
        v16bf bm0 = loadB(buf, 0);
        v16bf bm1 = loadB(buf, 1);
        v16bf bm2 = loadB(buf, 2);
        v16bf bm3 = loadB(buf, 3);

        acc[0][0] = __builtin_amdgcn_wmma_f32_16x16x32_bf16(false, a0, false, bm0, (short)0, acc[0][0], false, false);
        acc[1][0] = __builtin_amdgcn_wmma_f32_16x16x32_bf16(false, a1, false, bm0, (short)0, acc[1][0], false, false);
        acc[2][0] = __builtin_amdgcn_wmma_f32_16x16x32_bf16(false, a2, false, bm0, (short)0, acc[2][0], false, false);
        acc[3][0] = __builtin_amdgcn_wmma_f32_16x16x32_bf16(false, a3, false, bm0, (short)0, acc[3][0], false, false);
        acc[0][1] = __builtin_amdgcn_wmma_f32_16x16x32_bf16(false, a0, false, bm1, (short)0, acc[0][1], false, false);
        acc[1][1] = __builtin_amdgcn_wmma_f32_16x16x32_bf16(false, a1, false, bm1, (short)0, acc[1][1], false, false);
        acc[2][1] = __builtin_amdgcn_wmma_f32_16x16x32_bf16(false, a2, false, bm1, (short)0, acc[2][1], false, false);
        acc[3][1] = __builtin_amdgcn_wmma_f32_16x16x32_bf16(false, a3, false, bm1, (short)0, acc[3][1], false, false);
        acc[0][2] = __builtin_amdgcn_wmma_f32_16x16x32_bf16(false, a0, false, bm2, (short)0, acc[0][2], false, false);
        acc[1][2] = __builtin_amdgcn_wmma_f32_16x16x32_bf16(false, a1, false, bm2, (short)0, acc[1][2], false, false);
        acc[2][2] = __builtin_amdgcn_wmma_f32_16x16x32_bf16(false, a2, false, bm2, (short)0, acc[2][2], false, false);
        acc[3][2] = __builtin_amdgcn_wmma_f32_16x16x32_bf16(false, a3, false, bm2, (short)0, acc[3][2], false, false);
        acc[0][3] = __builtin_amdgcn_wmma_f32_16x16x32_bf16(false, a0, false, bm3, (short)0, acc[0][3], false, false);
        acc[1][3] = __builtin_amdgcn_wmma_f32_16x16x32_bf16(false, a1, false, bm3, (short)0, acc[1][3], false, false);
        acc[2][3] = __builtin_amdgcn_wmma_f32_16x16x32_bf16(false, a2, false, bm3, (short)0, acc[2][3], false, false);
        acc[3][3] = __builtin_amdgcn_wmma_f32_16x16x32_bf16(false, a3, false, bm3, (short)0, acc[3][3], false, false);

        wait_async0();      // our fills of buf^1 have landed in LDS
        __syncthreads();    // everyone's fills have landed
    }

    // ---- fused LSTM epilogue ----
    const int ug = uBase + waveN * 16 + lidx;
    const float bi = b_i[ug];
    const float bf = b_f[ug];
    const float bc = b_c[ug];
    const float bo = b_o[ug];

    #pragma unroll
    for (int mt = 0; mt < 4; ++mt) {
        const int row0 = blockRow + waveM * 64 + mt * 16 + half * 8;
        #pragma unroll
        for (int r = 0; r < 8; ++r) {
            const int row = row0 + r;
            const size_t idx = (size_t)row * UDIM + ug;
            const float zi = acc[mt][0][r] + bi;
            const float zf = acc[mt][1][r] + bf;
            const float zc = acc[mt][2][r] + bc;
            const float zo = acc[mt][3][r] + bo;
            const float it = fast_sigmoid(zi);
            const float ft = fast_sigmoid(zf);
            const float ct = fast_tanh(zc);
            const float ot = fast_sigmoid(zo);
            const float cold = Cin[idx];
            const float cnew = ft * cold + it * ct;
            const float hnew = ot * fast_tanh(cnew);
            Hout[idx] = hnew;
            Cout[idx] = cnew;
        }
    }
}

// ---------------------------------------------------------------------------
// Fallback (no workspace): fp32->bf16 conversion inline, BM=64 tile.
// ---------------------------------------------------------------------------
__global__ __launch_bounds__(256)
void lstm_wmma_kernel(const float* __restrict__ X,   const float* __restrict__ H,
                      const float* __restrict__ Cin,
                      const float* __restrict__ w_i, const float* __restrict__ u_i,
                      const float* __restrict__ w_f, const float* __restrict__ u_f,
                      const float* __restrict__ w_c, const float* __restrict__ u_c,
                      const float* __restrict__ w_o, const float* __restrict__ u_o,
                      const float* __restrict__ b_i, const float* __restrict__ b_f,
                      const float* __restrict__ b_c, const float* __restrict__ b_o,
                      float* __restrict__ Hout, float* __restrict__ Cout)
{
    const int tid   = threadIdx.x;
    const int wid   = tid >> 5;
    const int lane  = tid & 31;
    const int waveM = wid >> 2;
    const int waveN = wid & 3;
    const int half  = lane >> 4;
    const int lidx  = lane & 15;
    const int blockRow = blockIdx.y * FBM;
    const int uBase    = blockIdx.x * BN;

    __shared__ __align__(16) __bf16 sA[2][FBM][ASTRIDE];
    __shared__ __align__(16) __bf16 sB[2][NZ][BSTRIDE];

    auto stageA = [&](int buf, int ks) {
        const int row  = tid >> 2;
        const int q0   = tid & 3;
        const int kBase = ks * BK;
        const float* base = (kBase < DDIM)
            ? (X + (size_t)(blockRow + row) * DDIM + kBase)
            : (H + (size_t)(blockRow + row) * UDIM + (kBase - DDIM));
        #pragma unroll
        for (int qq = 0; qq < 2; ++qq) {
            const int q = q0 + qq * 4;
            v4f f = *(const v4f*)(base + q * 4);
            v4bf b;
            b.x = (__bf16)f.x; b.y = (__bf16)f.y; b.z = (__bf16)f.z; b.w = (__bf16)f.w;
            *(v4bf*)&sA[buf][row][q * 4] = b;
        }
    };
    auto stageB = [&](int buf, int ks) {
        #pragma unroll
        for (int pass = 0; pass < 4; ++pass) {
            const int task = pass * 256 + tid;
            const int uq = task & 63;
            const int kp = task >> 6;
            const int c0 = uq * 4;
            const int g  = c0 >> 6;
            const int ul = c0 & 63;
            const int k0 = ks * BK + kp * 2;
            const float* gptr;
            if (k0 < DDIM) {
                gptr = (g == 0) ? w_i : (g == 1) ? w_f : (g == 2) ? w_c : w_o;
                gptr += (size_t)k0 * UDIM;
            } else {
                gptr = (g == 0) ? u_i : (g == 1) ? u_f : (g == 2) ? u_c : u_o;
                gptr += (size_t)(k0 - DDIM) * UDIM;
            }
            const float* p = gptr + uBase + ul;
            v4f fa = *(const v4f*)p;
            v4f fb = *(const v4f*)(p + UDIM);
            #pragma unroll
            for (int i = 0; i < 4; ++i) {
                v2bf pk;
                pk.x = (__bf16)fa[i];
                pk.y = (__bf16)fb[i];
                *(v2bf*)&sB[buf][c0 + i][kp * 2] = pk;
            }
        }
    };
    auto loadA = [&](int buf, int mt) -> v16bf {
        const int row = waveM * 32 + mt * 16 + lidx;
        const __bf16* p = &sA[buf][row][half * 8];
        v8bf lo = *(const v8bf*)p;
        v8bf hi = *(const v8bf*)(p + 16);
        return __builtin_shufflevector(lo, hi, 0,1,2,3,4,5,6,7,8,9,10,11,12,13,14,15);
    };
    auto loadB = [&](int buf, int g) -> v16bf {
        const int row = g * 64 + waveN * 16 + lidx;
        const __bf16* p = &sB[buf][row][half * 8];
        v8bf lo = *(const v8bf*)p;
        v8bf hi = *(const v8bf*)(p + 16);
        return __builtin_shufflevector(lo, hi, 0,1,2,3,4,5,6,7,8,9,10,11,12,13,14,15);
    };

    v8f acc[2][4] = {};
    stageA(0, 0);
    stageB(0, 0);
    __syncthreads();
    const int NK = KTOT / BK;
    for (int ks = 0; ks < NK; ++ks) {
        const int buf = ks & 1;
        if (ks + 1 < NK) { stageA(buf ^ 1, ks + 1); stageB(buf ^ 1, ks + 1); }
        v16bf a0 = loadA(buf, 0);
        v16bf a1 = loadA(buf, 1);
        #pragma unroll
        for (int g = 0; g < 4; ++g) {
            v16bf bm = loadB(buf, g);
            acc[0][g] = __builtin_amdgcn_wmma_f32_16x16x32_bf16(
                false, a0, false, bm, (short)0, acc[0][g], false, false);
            acc[1][g] = __builtin_amdgcn_wmma_f32_16x16x32_bf16(
                false, a1, false, bm, (short)0, acc[1][g], false, false);
        }
        __syncthreads();
    }

    const int ug = uBase + waveN * 16 + lidx;
    const float bi = b_i[ug];
    const float bfv = b_f[ug];
    const float bc = b_c[ug];
    const float bo = b_o[ug];
    #pragma unroll
    for (int mt = 0; mt < 2; ++mt) {
        const int row0 = blockRow + waveM * 32 + mt * 16 + half * 8;
        #pragma unroll
        for (int r = 0; r < 8; ++r) {
            const int row = row0 + r;
            const size_t idx = (size_t)row * UDIM + ug;
            const float zi = acc[mt][0][r] + bi;
            const float zf = acc[mt][1][r] + bfv;
            const float zc = acc[mt][2][r] + bc;
            const float zo = acc[mt][3][r] + bo;
            const float it = fast_sigmoid(zi);
            const float ft = fast_sigmoid(zf);
            const float ct = fast_tanh(zc);
            const float ot = fast_sigmoid(zo);
            const float cold = Cin[idx];
            const float cnew = ft * cold + it * ct;
            const float hnew = ot * fast_tanh(cnew);
            Hout[idx] = hnew;
            Cout[idx] = cnew;
        }
    }
}

extern "C" void kernel_launch(void* const* d_in, const int* in_sizes, int n_in,
                              void* d_out, int out_size, void* d_ws, size_t ws_size,
                              hipStream_t stream) {
    const float* X   = (const float*)d_in[0];
    const float* H   = (const float*)d_in[1];
    const float* C   = (const float*)d_in[2];
    const float* w_i = (const float*)d_in[3];
    const float* u_i = (const float*)d_in[4];
    const float* w_f = (const float*)d_in[5];
    const float* u_f = (const float*)d_in[6];
    const float* w_c = (const float*)d_in[7];
    const float* u_c = (const float*)d_in[8];
    const float* w_o = (const float*)d_in[9];
    const float* u_o = (const float*)d_in[10];
    const float* b_i = (const float*)d_in[11];
    const float* b_f = (const float*)d_in[12];
    const float* b_c = (const float*)d_in[13];
    const float* b_o = (const float*)d_in[14];

    float* Hout = (float*)d_out;
    float* Cout = Hout + (size_t)B_ROWS * UDIM;

    const size_t needX = (size_t)B_ROWS * KTOT * sizeof(__hip_bfloat16);
    const size_t needW = (size_t)4 * UDIM * KTOT * sizeof(__hip_bfloat16);

    if (d_ws != nullptr && ws_size >= needX + needW) {
        __bf16* Xbf = (__bf16*)d_ws;
        __bf16* Wbf = (__bf16*)((char*)d_ws + needX);
        pack_x_kernel<<<(B_ROWS * (KTOT / 4)) / 256, 256, 0, stream>>>(X, H, Xbf);
        pack_w_kernel<<<(4 * UDIM * (KTOT / 4)) / 256, 256, 0, stream>>>(
            w_i, u_i, w_f, u_f, w_c, u_c, w_o, u_o, Wbf);
        dim3 grid(UDIM / BN, B_ROWS / BM);   // (8, 128)
        lstm_wmma_async_kernel<<<grid, 256, 0, stream>>>(
            Xbf, Wbf, C, b_i, b_f, b_c, b_o, Hout, Cout);
    } else {
        dim3 grid(UDIM / BN, B_ROWS / FBM);  // (8, 256)
        lstm_wmma_kernel<<<grid, 256, 0, stream>>>(
            X, H, C, w_i, u_i, w_f, u_f, w_c, u_c, w_o, u_o,
            b_i, b_f, b_c, b_o, Hout, Cout);
    }
}
